// CrossAttention_2327872274452
// MI455X (gfx1250) — matrix operand order repaired and checked by reading, not compiled, not run on hardware
//
#include <hip/hip_runtime.h>

#define QD  1024
#define KVD 768
#define NH  16
#define HD  64
#define BB  4
#define SS  2048

typedef __attribute__((ext_vector_type(16))) __bf16 v16bf;
typedef __attribute__((ext_vector_type(8)))  float  v8f;

static __device__ __forceinline__ unsigned short f32_to_bf16_bits(float f) {
  unsigned u = __builtin_bit_cast(unsigned, f);
  u += 0x7fffu + ((u >> 16) & 1u);          // round-to-nearest-even
  return (unsigned short)(u >> 16);
}
static __device__ __forceinline__ __bf16 f32_to_bf16(float f) {
  unsigned short s = f32_to_bf16_bits(f);
  return __builtin_bit_cast(__bf16, s);
}

// ---------------------------------------------------------------------------
// Elementwise f32 -> bf16 convert (activations), 4 elements per thread.
// ---------------------------------------------------------------------------
__global__ void pack_x(const float* __restrict__ X, unsigned short* __restrict__ Xb, int n) {
  int i0 = (blockIdx.x * blockDim.x + threadIdx.x) * 4;
  if (i0 + 3 < n) {
#pragma unroll
    for (int j = 0; j < 4; ++j) Xb[i0 + j] = f32_to_bf16_bits(X[i0 + j]);
  } else {
    for (int j = 0; j < 4 && i0 + j < n; ++j) Xb[i0 + j] = f32_to_bf16_bits(X[i0 + j]);
  }
}

// ---------------------------------------------------------------------------
// Pack weights: W [Kd x 1024] f32 (row-major)  ->  Wt [1024 x Kd] bf16
// (transposed so WMMA B-matrix loads are contiguous 32B per lane)
// ---------------------------------------------------------------------------
__global__ void pack_w(const float* __restrict__ W, unsigned short* __restrict__ Wt, int Kd) {
  int idx = blockIdx.x * blockDim.x + threadIdx.x;
  if (idx >= Kd * QD) return;
  int k = idx / QD, n = idx - k * QD;
  Wt[(size_t)n * Kd + k] = f32_to_bf16_bits(W[idx]);
}

// ---------------------------------------------------------------------------
// Projection GEMM: C[M,1024] = X[M,Kd] @ W[Kd,1024] + bias   (X, W in bf16)
// One wave computes a 16x128 tile (8 col-tiles, 8 WMMAs per k-step;
// 64 accumulator VGPRs -> no spills).
// mode 0: out bf16  [B,H,S,D]   (Q / K head-major)
// mode 1: out bf16  [B,H,D,S]   (V transposed)
// mode 2: out f32   [M,1024]    (final output)
// ---------------------------------------------------------------------------
__global__ void proj_gemm(const unsigned short* __restrict__ Xw,
                          const unsigned short* __restrict__ Wtb,
                          const float* __restrict__ bias,
                          void* __restrict__ outv, int M, int Kd, int mode) {
  const int lane = threadIdx.x & 31;
  const int hx   = lane >> 4;                 // wave half (0/1)
  const int col  = lane & 15;
  const int wave = (blockIdx.x * blockDim.x + threadIdx.x) >> 5;
  const int tiles_n = QD / 128;               // 8
  const int mt = wave / tiles_n;
  const int nt = wave - mt * tiles_n;
  if (mt >= M / 16) return;
  const int m0 = mt * 16, n0 = nt * 128;

  const __bf16* Xb = (const __bf16*)Xw;
  const __bf16* Wt = (const __bf16*)Wtb;

  v8f acc[8] = {};
  const size_t xrow = (size_t)(m0 + col) * Kd;

  for (int k0 = 0; k0 < Kd; k0 += 32) {
    // A tile (16x32): lane row = col, K = 16*half + j
    v16bf a;
    const __bf16* p0 = Xb + xrow + k0 + 16 * hx;
#pragma unroll
    for (int j = 0; j < 8; ++j) { a[j] = p0[j]; a[j + 8] = p0[j + 8]; }

#pragma unroll
    for (int t = 0; t < 8; ++t) {
      // B tile (32x16): lane col = n, K = 16*half + j -> contiguous 32B of Wt[n]
      const int n = n0 + t * 16 + col;
      v16bf bm = *(const v16bf*)(Wt + (size_t)n * Kd + k0 + 16 * hx);
      acc[t] = __builtin_amdgcn_wmma_f32_16x16x32_bf16(false, a, false, bm,
                                                       (short)0, acc[t], false, false);
    }
  }

#pragma unroll
  for (int t = 0; t < 8; ++t) {
    const int n = n0 + t * 16 + col;
    const float bv = bias[n];
#pragma unroll
    for (int r = 0; r < 8; ++r) {
      const int m = m0 + r + 8 * hx;            // C layout rows r / r+8 per half
      const float val = acc[t][r] + bv;
      if (mode == 2) {
        ((float*)outv)[(size_t)m * QD + n] = val;
      } else {
        const int bb = m / SS, s = m - bb * SS;
        const int hh = n >> 6, dd = n & 63;
        size_t idx = (mode == 0)
          ? (((size_t)(bb * NH + hh) * SS + s) * HD + dd)     // [B,H,S,D]
          : (((size_t)(bb * NH + hh) * HD + dd) * SS + s);    // [B,H,D,S]
        ((unsigned short*)outv)[idx] = f32_to_bf16_bits(val);
      }
    }
  }
}

// ---------------------------------------------------------------------------
// Flash attention: one wave handles (b, h, 16-row q-tile), streams KV in
// blocks of 32 with online softmax.  Computes S^T = K_blk @ Q^T so each lane
// owns a single q column (softmax reduction = 16 in-lane values + shfl_xor 16),
// then O^T += V^T @ P^T.  All GEMMs via v_wmma_f32_16x16x32_bf16.
// ---------------------------------------------------------------------------
__global__ void attn_fa(const unsigned short* __restrict__ Qw,
                        const unsigned short* __restrict__ Kw,
                        const unsigned short* __restrict__ Vw,
                        unsigned short* __restrict__ ctx) {
  const int lane = threadIdx.x & 31;
  const int hx   = lane >> 4;
  const int col  = lane & 15;
  const int wave = (blockIdx.x * blockDim.x + threadIdx.x) >> 5;
  const int qtiles = SS / 16;                 // 128
  const int qt = wave % qtiles;
  const int h  = (wave / qtiles) % NH;
  const int b  = wave / (qtiles * NH);
  if (b >= BB) return;
  const int q0 = qt * 16;

  const size_t bh = (size_t)(b * NH + h);
  const __bf16* Qh = (const __bf16*)Qw + bh * SS * HD;   // [S, D]
  const __bf16* Kh = (const __bf16*)Kw + bh * SS * HD;   // [S, D]
  const __bf16* Vh = (const __bf16*)Vw + bh * HD * SS;   // [D, S]

  // Q^T as B-matrix (K = d, N = q), reused for all KV blocks; two 32-d chunks
  v16bf bq[2];
#pragma unroll
  for (int dc = 0; dc < 2; ++dc)
    bq[dc] = *(const v16bf*)(Qh + (size_t)(q0 + col) * HD + dc * 32 + 16 * hx);

  v8f o[4] = {};                              // O^T accum: 4 tiles of 16 d-rows
  float mrun = -3.0e38f, lrun = 0.0f;

  for (int kv0 = 0; kv0 < SS; kv0 += 32) {
    // ---- S^T = K_blk @ Q^T : two 16(kv) x 16(q) f32 tiles ----
    v8f st[2] = {};
#pragma unroll
    for (int sub = 0; sub < 2; ++sub) {
      const __bf16* krow = Kh + (size_t)(kv0 + 16 * sub + col) * HD;
#pragma unroll
      for (int dc = 0; dc < 2; ++dc) {
        v16bf a;                              // A: lane row = kv, K = 16*half + j
        const __bf16* p0 = krow + dc * 32 + 16 * hx;
#pragma unroll
        for (int j = 0; j < 8; ++j) { a[j] = p0[j]; a[j + 8] = p0[j + 8]; }
        st[sub] = __builtin_amdgcn_wmma_f32_16x16x32_bf16(false, a, false, bq[dc],
                                                          (short)0, st[sub], false, false);
      }
    }

    // ---- online softmax (per lane = one q column; kv split across halves) ----
    float mloc = -3.0e38f;
#pragma unroll
    for (int r = 0; r < 8; ++r) {
      mloc = fmaxf(mloc, st[0][r]);
      mloc = fmaxf(mloc, st[1][r]);
    }
    mloc = fmaxf(mloc, __shfl_xor(mloc, 16, 32));
    const float mnew  = fmaxf(mrun, mloc);
    const float alpha = __expf(0.125f * (mrun - mnew));   // 0.125 = 1/sqrt(D)
    float p0a[8], p1a[8], ssum = 0.0f;
#pragma unroll
    for (int r = 0; r < 8; ++r) {
      p0a[r] = __expf(0.125f * (st[0][r] - mnew));
      p1a[r] = __expf(0.125f * (st[1][r] - mnew));
      ssum += p0a[r] + p1a[r];
    }
    lrun = lrun * alpha + ssum + __shfl_xor(ssum, 16, 32);
    mrun = mnew;
#pragma unroll
    for (int t = 0; t < 4; ++t)
#pragma unroll
      for (int r = 0; r < 8; ++r) o[t][r] *= alpha;

    // ---- P^T as B-matrix (K = kv in 0..31, N = q): half-swap via shfl_xor ----
    v16bf pb;
#pragma unroll
    for (int r = 0; r < 8; ++r) {
      const float o0 = __shfl_xor(p0a[r], 16, 32);
      const float o1 = __shfl_xor(p1a[r], 16, 32);
      pb[r]     = f32_to_bf16(hx ? o1     : p0a[r]);  // K = r      (half0) / 16+r (half1)
      pb[r + 8] = f32_to_bf16(hx ? p1a[r] : o0);      // K = 8+r    (half0) / 24+r (half1)
    }

    // ---- O^T += V^T @ P^T : A = V^T (lane row = d, contiguous kv chunks) ----
#pragma unroll
    for (int t = 0; t < 4; ++t) {
      const __bf16* vrow = Vh + (size_t)(t * 16 + col) * SS + kv0 + 16 * hx;
      v16bf a;
#pragma unroll
      for (int j = 0; j < 8; ++j) { a[j] = vrow[j]; a[j + 8] = vrow[j + 8]; }
      o[t] = __builtin_amdgcn_wmma_f32_16x16x32_bf16(false, a, false, pb,
                                                     (short)0, o[t], false, false);
    }
  }

  // ---- normalize and write ctx bf16 in [B, S, H*D] for the output GEMM ----
  const float rinv = 1.0f / lrun;
  unsigned short* crow = ctx + (size_t)(b * SS + q0 + col) * QD + h * HD;
#pragma unroll
  for (int t = 0; t < 4; ++t)
#pragma unroll
    for (int r = 0; r < 8; ++r)
      crow[t * 16 + r + 8 * hx] = f32_to_bf16_bits(o[t][r] * rinv);
}

// ---------------------------------------------------------------------------
extern "C" void kernel_launch(void* const* d_in, const int* in_sizes, int n_in,
                              void* d_out, int out_size, void* d_ws, size_t ws_size,
                              hipStream_t stream) {
  (void)in_sizes; (void)n_in; (void)out_size; (void)ws_size;
  const float* query = (const float*)d_in[0];
  const float* key   = (const float*)d_in[1];
  const float* value = (const float*)d_in[2];
  const float* Wq = (const float*)d_in[3];
  const float* bq = (const float*)d_in[4];
  const float* Wk = (const float*)d_in[5];
  const float* bk = (const float*)d_in[6];
  const float* Wv = (const float*)d_in[7];
  const float* bv = (const float*)d_in[8];
  const float* Wo = (const float*)d_in[9];
  const float* bo = (const float*)d_in[10];

  char* ws = (char*)d_ws;
  size_t off = 0;
  auto alloc = [&](size_t bytes) -> void* {
    void* p = ws + off;
    off = (off + bytes + 255) & ~(size_t)255;
    return p;
  };
  const size_t act = (size_t)BB * SS * QD;                  // 8.4M elems
  const size_t akv = (size_t)BB * SS * KVD;                 // 6.3M elems
  unsigned short* Qbf = (unsigned short*)alloc(act * 2);    // [B,H,S,D]
  unsigned short* Kbf = (unsigned short*)alloc(act * 2);    // [B,H,S,D]
  unsigned short* Vtb = (unsigned short*)alloc(act * 2);    // [B,H,D,S]
  unsigned short* ctx = (unsigned short*)alloc(act * 2);    // [B,S,H*D]
  unsigned short* Xq  = (unsigned short*)alloc(act * 2);    // bf16 query
  unsigned short* Xk  = (unsigned short*)alloc(akv * 2);    // bf16 key
  unsigned short* Xv  = (unsigned short*)alloc(akv * 2);    // bf16 value
  unsigned short* Wqt = (unsigned short*)alloc((size_t)QD  * QD * 2);
  unsigned short* Wkt = (unsigned short*)alloc((size_t)KVD * QD * 2);
  unsigned short* Wvt = (unsigned short*)alloc((size_t)KVD * QD * 2);
  unsigned short* Wot = (unsigned short*)alloc((size_t)QD  * QD * 2);

  // 1) bf16-pack activations and (transposed) weights
  pack_x<<<(int)((act + 1023) / 1024), 256, 0, stream>>>(query, Xq, (int)act);
  pack_x<<<(int)((akv + 1023) / 1024), 256, 0, stream>>>(key,   Xk, (int)akv);
  pack_x<<<(int)((akv + 1023) / 1024), 256, 0, stream>>>(value, Xv, (int)akv);
  const int nq = QD * QD, nkv = KVD * QD;
  pack_w<<<(nq  + 255) / 256, 256, 0, stream>>>(Wq, Wqt, QD);
  pack_w<<<(nkv + 255) / 256, 256, 0, stream>>>(Wk, Wkt, KVD);
  pack_w<<<(nkv + 255) / 256, 256, 0, stream>>>(Wv, Wvt, KVD);
  pack_w<<<(nq  + 255) / 256, 256, 0, stream>>>(Wo, Wot, QD);

  // 2) Q/K/V projections (WMMA, 16x128 tile per wave)
  const int M = BB * SS;                                    // 8192
  const int wavesProj = (M / 16) * (QD / 128);              // 4096 waves
  const int blkProj = wavesProj / 8;                        // 256 thr = 8 waves
  proj_gemm<<<blkProj, 256, 0, stream>>>(Xq, Wqt, bq, Qbf, M, QD,  0);
  proj_gemm<<<blkProj, 256, 0, stream>>>(Xk, Wkt, bk, Kbf, M, KVD, 0);
  proj_gemm<<<blkProj, 256, 0, stream>>>(Xv, Wvt, bv, Vtb, M, KVD, 1);

  // 3) flash attention (WMMA, online softmax, 16 q per wave)
  const int wavesAttn = BB * NH * (SS / 16);                // 8192 waves
  attn_fa<<<wavesAttn / 8, 256, 0, stream>>>(Qbf, Kbf, Vtb, ctx);

  // 4) output projection -> f32 d_out
  proj_gemm<<<blkProj, 256, 0, stream>>>(ctx, Wot, bo, d_out, M, QD, 2);
}
